// EvoformerBlock_29240137351805
// MI455X (gfx1250) — matrix operand order repaired and checked by reading, not compiled, not run on hardware
//
// Evoformer block for MI455X (gfx1250, wave32, WMMA + TDM). Compile-only target.
// GEMMs: v_wmma_f32_16x16x32_bf16 (bf16 operands, f32 accumulate), tiles staged
// into LDS by the Tensor Data Mover (tensor_load_to_lds + s_wait_tensorcnt),
// with DMA-side bank padding and hardware OOB zero-fill for edge N tiles.
// Workspace layout assumes ws_size >= ~350 MB (6 pair-sized slabs + small).
#include <hip/hip_runtime.h>
#include <hip/hip_bf16.h>

typedef __attribute__((ext_vector_type(16))) __bf16          v16bf;
typedef __attribute__((ext_vector_type(8)))  float           v8f;
typedef __attribute__((ext_vector_type(16))) unsigned short  v16us;
typedef __attribute__((ext_vector_type(4)))  unsigned int    v4ui;
typedef __attribute__((ext_vector_type(8)))  int             v8i;
typedef __attribute__((ext_vector_type(4)))  int             v4i;

#if defined(__HIP_DEVICE_COMPILE__) && defined(__gfx1250__) && defined(__has_builtin)
#  if __has_builtin(__builtin_amdgcn_tensor_load_to_lds)
#    define HAVE_TDM 1
#  else
#    define HAVE_TDM 0
#  endif
#else
#  define HAVE_TDM 0
#endif

#define Lc   320
#define DIMc 256
#define PD   128
#define Hc   8
#define HPc  4
#define FFNc 1024
#define TOK  (Lc * Lc)        /* 102400 pair tokens */
#define CI   32               /* pair-attention i-chunk */
#define MC   25600            /* pFFN token chunk */

__device__ __forceinline__ unsigned short f2bf(float f) {
  unsigned u = __float_as_uint(f);
  u += 0x7FFFu + ((u >> 16) & 1u);          // round-to-nearest-even bf16
  return (unsigned short)(u >> 16);
}
__device__ __forceinline__ float gelu_exact(float x) {
  return 0.5f * x * (1.0f + erff(x * 0.70710678118654752f));
}
__device__ __forceinline__ float sigmoidf_(float x) {
  return 1.0f / (1.0f + __expf(-x));
}

#if HAVE_TDM
// TDM: DMA one 64(row) x 32(elem) f32 tile, row stride `stride` elements,
// into LDS at byte address lds_addr with a 1-dword pad every 32 dwords
// (=> 33-float LDS row stride, kills DS bank conflicts on fragment reads).
// rows/cols beyond (rows_avail, cols_avail) zero-fill via D# tensor dims.
__device__ __forceinline__ void tdm_load_tile_f32(unsigned lds_addr,
                                                  const float* gp,
                                                  long stride,
                                                  int rows_avail,
                                                  int cols_avail) {
  unsigned long long ga = (unsigned long long)(size_t)gp;
  v4ui g0;
  g0[0] = 1u;                                              // count=1 (user D#)
  g0[1] = lds_addr;                                        // lds_addr
  g0[2] = (unsigned)(ga & 0xFFFFFFFFull);                  // global_addr[31:0]
  g0[3] = (unsigned)((ga >> 32) & 0x1FFFFFFull) | (2u << 30); // [56:32] | type=2
  const unsigned td0 = (unsigned)cols_avail;               // tensor_dim0
  const unsigned td1 = (unsigned)rows_avail;               // tensor_dim1
  const unsigned long long s0 = (unsigned long long)stride;
  v8i g1;
  g1[0] = (int)((2u << 16) | (1u << 20) | (4u << 22));     // 4B elems, pad 1dw/32dw
  g1[1] = (int)((td0 & 0xFFFFu) << 16);
  g1[2] = (int)((td0 >> 16) | ((td1 & 0xFFFFu) << 16));
  g1[3] = (int)((td1 >> 16) | (32u << 16));                // tile_dim0 = 32
  g1[4] = (int)(64u);                                      // tile_dim1 = 64
  g1[5] = (int)(s0 & 0xFFFFFFFFull);
  g1[6] = (int)((unsigned)(s0 >> 32) & 0xFFFFu);           // dim0_stride[47:32]
  g1[7] = 0;
  v4i z4 = {0, 0, 0, 0};
#if __clang_major__ >= 23
  v8i z8 = {0, 0, 0, 0, 0, 0, 0, 0};
  __builtin_amdgcn_tensor_load_to_lds(g0, g1, z4, z4, z8, 0);
#else
  __builtin_amdgcn_tensor_load_to_lds(g0, g1, z4, z4, 0);
#endif
}
#endif

// ---------------------------------------------------------------------------
// Generic strided-batched GEMM:  Y = epi( alpha * X[M,K] @ op(W) + bias )
//   BT=true : op(W) = W[N,K]^T (torch Linear weight)   BT=false: W is [K,N]
//   epi: ACT (0=none, 1=exact gelu), optional *= sigmoid(*gate), += residual
//   batch z -> (zo = z/nbi, zi = z%nbi) with independent offsets per operand.
// Block tile 64Mx64N, 8 waves as 4x2, each wave 16x32 (2 WMMA per K-step).
// Requires M % 64 == 0 and K % 32 == 0 (true for every call site here).
// ---------------------------------------------------------------------------
struct GemmArgs {
  const float *X, *W, *bias, *R, *gate;
  float *Y;
  long xs, ws;                  // row strides of X and W
  long ys_m, ys_n, rs_m, rs_n;  // output / residual strides
  int  M, N, K;
  float alpha;
  int  nbi;                     // inner batch modulus
  long xb_o, xb_i, wb_o, wb_i, yb_o, yb_i, rb_o, rb_i;
};

template <int ACT, bool BT>
__global__ __launch_bounds__(256) void k_gemm(GemmArgs g) {
  __shared__ float lAf[64][33];           // 64(M) x 32(K) f32, 33-f row stride
  __shared__ float lBf[64][33];           // 64(N) x 32(K) f32
  const int tid  = threadIdx.x;
  const int lane = tid & 31;
  const int w    = tid >> 5;
  const int wm   = w >> 1, wn = w & 1;    // 4x2 wave tiling of 64x64
  const int m0   = blockIdx.y * 64;
  const int n0   = blockIdx.x * 64;
  const int z    = blockIdx.z;
  const int zo   = z / g.nbi, zi = z % g.nbi;
  const float* X = g.X + (long)zo * g.xb_o + (long)zi * g.xb_i;
  const float* W = g.W + (long)zo * g.wb_o + (long)zi * g.wb_i;
  float*       Y = g.Y + (long)zo * g.yb_o + (long)zi * g.yb_i;
  const float* R = g.R ? g.R + (long)zo * g.rb_o + (long)zi * g.rb_i : nullptr;

  v8f acc[2] = {};
  for (int k0 = 0; k0 < g.K; k0 += 32) {
#if HAVE_TDM
    // --- async tile staging via the Tensor Data Mover ---
    if (w == 0) {
      tdm_load_tile_f32((unsigned)(size_t)&lAf[0][0],
                        &X[(long)m0 * g.xs + k0], g.xs, g.M - m0, g.K - k0);
      __builtin_amdgcn_s_wait_tensorcnt(0);
    }
    if (BT) {
      if (w == 1) {
        tdm_load_tile_f32((unsigned)(size_t)&lBf[0][0],
                          &W[(long)n0 * g.ws + k0], g.ws, g.N - n0, g.K - k0);
        __builtin_amdgcn_s_wait_tensorcnt(0);
      }
    }
#else
    // --- cooperative fallback: vectorized float4 loads ---
#pragma unroll
    for (int r = 0; r < 2; ++r) {
      int v = tid + 256 * r;                          // 512 float4 per tile
      int mm = v >> 3, kq = (v & 7) * 4;
      const float4 x4 = *(const float4*)&X[(long)(m0 + mm) * g.xs + k0 + kq];
      lAf[mm][kq + 0] = x4.x; lAf[mm][kq + 1] = x4.y;
      lAf[mm][kq + 2] = x4.z; lAf[mm][kq + 3] = x4.w;
    }
    if (BT) {
#pragma unroll
      for (int r = 0; r < 2; ++r) {
        int v = tid + 256 * r;
        int nn = v >> 3, kq = (v & 7) * 4;
        float4 w4 = {0.f, 0.f, 0.f, 0.f};
        if (n0 + nn < g.N)
          w4 = *(const float4*)&W[(long)(n0 + nn) * g.ws + k0 + kq];
        lBf[nn][kq + 0] = w4.x; lBf[nn][kq + 1] = w4.y;
        lBf[nn][kq + 2] = w4.z; lBf[nn][kq + 3] = w4.w;
      }
    }
#endif
    if (!BT) {
      // W is [K][N]: transpose into LDS [n][k]; n fastest for coalescing.
#pragma unroll
      for (int r = 0; r < 8; ++r) {
        int p = tid + 256 * r;
        int nn = p & 63, kk = p >> 6;
        float v = 0.f;
        if (n0 + nn < g.N) v = W[(long)(k0 + kk) * g.ws + n0 + nn];
        lBf[nn][kk] = v;
      }
    }
    __syncthreads();
    if (k0 + 32 < g.K)                                // gfx1250 global_prefetch_b8
      __builtin_prefetch(&X[(long)(m0 + w * 8) * g.xs + k0 + 32], 0, 1);

    // Build wave32 WMMA fragments, k = 16*(lane/16) + e in A and B; cvt f32->bf16 here
    // (co-executes with the matrix pipe; TDM moved raw f32 bytes).
    const int ml = lane & 15;
    const int ko = (lane >> 4) * 16;                  // A: K sub-offset
    const int kb = (lane >> 4) * 16;                  // B: K half select
    const float* ar = lAf[wm * 16 + ml];
    v16us au;
#pragma unroll
    for (int e = 0; e < 8; ++e) {
      au[e]     = f2bf(ar[ko + e]);
      au[8 + e] = f2bf(ar[ko + 8 + e]);
    }
#pragma unroll
    for (int t = 0; t < 2; ++t) {
      const float* br = lBf[wn * 32 + t * 16 + ml];
      v16us bu;
#pragma unroll
      for (int e = 0; e < 16; ++e) bu[e] = f2bf(br[kb + e]);
      acc[t] = __builtin_amdgcn_wmma_f32_16x16x32_bf16(
          false, __builtin_bit_cast(v16bf, au),
          false, __builtin_bit_cast(v16bf, bu),
          (short)0, acc[t], false, false);
    }
    __syncthreads();
  }

  const float gt   = g.gate ? sigmoidf_(g.gate[0]) : 1.f;
  const int   mtop = m0 + wm * 16 + ((lane >> 4) ? 8 : 0);
#pragma unroll
  for (int t = 0; t < 2; ++t) {
    const int ng = n0 + wn * 32 + t * 16 + (lane & 15);
    if (ng >= g.N) continue;
    const float bs = g.bias ? g.bias[ng] : 0.f;
#pragma unroll
    for (int r = 0; r < 8; ++r) {                     // C/D: VGPR r <-> M=r / r+8
      const int mg = mtop + r;
      float v = g.alpha * acc[t][r] + bs;
      if (ACT == 1) v = gelu_exact(v);
      v *= gt;
      if (R) v += R[(long)mg * g.rs_m + (long)ng * g.rs_n];
      Y[(long)mg * g.ys_m + (long)ng * g.ys_n] = v;
    }
  }
}

// ------------------------------ row primitives -----------------------------
// LayerNorm: one wave32 per row, D in {128,256}. Optional transposed store:
// row (r0,r1) of (rows/R1, R1) written to output row r1*(rows/R1)+r0.
__global__ void k_ln(const float* x, const float* g, const float* b, float* y,
                     long rows, int D, int R1, int tstore) {
  long gw  = (((long)blockIdx.x * blockDim.x) + threadIdx.x) >> 5;
  int lane = threadIdx.x & 31;
  if (gw >= rows) return;
  const float* xr = x + gw * D;
  const int per = D >> 5;            // <= 8
  float v[8], s = 0.f;
  for (int i = 0; i < per; ++i) { v[i] = xr[lane + 32 * i]; s += v[i]; }
  for (int m = 16; m; m >>= 1) s += __shfl_xor(s, m, 32);
  const float mean = s / D;
  float var = 0.f;
  for (int i = 0; i < per; ++i) { float d = v[i] - mean; var += d * d; }
  for (int m = 16; m; m >>= 1) var += __shfl_xor(var, m, 32);
  const float inv = rsqrtf(var / D + 1e-5f);
  long orow = gw;
  if (tstore) { long r0 = gw / R1, r1 = gw % R1; orow = r1 * (rows / R1) + r0; }
  float* yr = y + orow * D;
  for (int i = 0; i < per; ++i) {
    int d = lane + 32 * i;
    yr[d] = (v[i] - mean) * inv * g[d] + b[d];
  }
}

// In-place softmax over rows of length N (N % 32 == 0, N <= 512).
__global__ void k_softmax(float* x, long rows, int N) {
  long gw  = (((long)blockIdx.x * blockDim.x) + threadIdx.x) >> 5;
  int lane = threadIdx.x & 31;
  if (gw >= rows) return;
  float* xr = x + gw * N;
  const int per = N >> 5;            // 10 for N=320
  float v[16], mx = -1e30f;
  for (int i = 0; i < per; ++i) { v[i] = xr[lane + 32 * i]; mx = fmaxf(mx, v[i]); }
  for (int m = 16; m; m >>= 1) mx = fmaxf(mx, __shfl_xor(mx, m, 32));
  float s = 0.f;
  for (int i = 0; i < per; ++i) { v[i] = __expf(v[i] - mx); s += v[i]; }
  for (int m = 16; m; m >>= 1) s += __shfl_xor(s, m, 32);
  const float inv = 1.f / s;
  for (int i = 0; i < per; ++i) xr[lane + 32 * i] = v[i] * inv;
}

// out[gw] = dot(x[gw,:], w) + b   (one wave per row)
__global__ void k_rowdot(const float* x, const float* w, const float* b,
                         float* out, long rows, int D) {
  long gw  = (((long)blockIdx.x * blockDim.x) + threadIdx.x) >> 5;
  int lane = threadIdx.x & 31;
  if (gw >= rows) return;
  const float* xr = x + gw * D;
  float s = 0.f;
  for (int d = lane; d < D; d += 32) s += xr[d] * w[d];
  for (int m = 16; m; m >>= 1) s += __shfl_xor(s, m, 32);
  if (lane == 0) out[gw] = s + b[0];
}

// pooled[i,d] = sum_j wts[i,j] * zn[i,j,d]
__global__ void k_pool(const float* wts, const float* zn, float* out, int L, int D) {
  const int i = blockIdx.x, d = threadIdx.x;
  float acc = 0.f;
  for (int j = 0; j < L; ++j)
    acc = fmaf(wts[i * L + j], zn[((long)i * L + j) * D + d], acc);
  out[(long)i * D + d] = acc;
}

// Triangular einsum, d-vectorized (d contiguous, both operands L2-resident):
//  mode 0 (outgoing): o[p,q,d] = sum_t a[p,t,d]*b[t,q,d]
//  mode 1 (incoming): o[p,q,d] = sum_t a[p,t,d]*b[q,t,d]
__global__ void k_tri(const float* a, const float* b, float* o,
                      int L, int D, int mode) {
  const int p = blockIdx.x, q = blockIdx.y, d = threadIdx.x;
  float acc = 0.f;
  const float* ap = a + ((long)p * L) * D + d;
  for (int t = 0; t < L; ++t) {
    const float av = ap[(long)t * D];
    const float bv = mode ? b[((long)q * L + t) * D + d]
                          : b[((long)t * L + q) * D + d];
    acc = fmaf(av, bv, acc);
  }
  o[((long)p * L + q) * D + d] = acc;
}

// ------------------------------- elementwise -------------------------------
__global__ void k_mul_sigmoid(float* a, const float* g, long n) {
  for (long i = (long)blockIdx.x * blockDim.x + threadIdx.x; i < n;
       i += (long)gridDim.x * blockDim.x)
    a[i] *= sigmoidf_(g[i]);
}
__global__ void k_gate_add(float* dst, const float* src, const float* gsrc, long n) {
  for (long i = (long)blockIdx.x * blockDim.x + threadIdx.x; i < n;
       i += (long)gridDim.x * blockDim.x)
    dst[i] += src[i] * sigmoidf_(gsrc[i]);
}
__global__ void k_geglu(const float* xg, float* out, int M, int F) {
  long n = (long)M * F;
  for (long i = (long)blockIdx.x * blockDim.x + threadIdx.x; i < n;
       i += (long)gridDim.x * blockDim.x) {
    long m = i / F, f = i % F;
    out[i] = xg[m * 2 * F + f] * gelu_exact(xg[m * 2 * F + F + f]);
  }
}
// pu_in[i,j,d] = gelu(U[i,d] + V[j,d] + b1[d])   (factorized s2p first linear)
__global__ void k_outer_gelu(const float* U, const float* V, const float* b1,
                             float* out, int L, int D) {
  long n = (long)L * L * D;
  for (long idx = (long)blockIdx.x * blockDim.x + threadIdx.x; idx < n;
       idx += (long)gridDim.x * blockDim.x) {
    long i = idx / ((long)L * D), r = idx % ((long)L * D);
    long j = r / D, d = r % D;
    out[idx] = gelu_exact(U[i * D + d] + V[j * D + d] + b1[d]);
  }
}
// P[i, j0+jl, d] += tmp[jl, i, d]   (column-attention residual untranspose)
__global__ void k_addT(float* P, const float* tmp, int j0, int CIn, int L, int D) {
  long n = (long)CIn * L * D;
  for (long idx = (long)blockIdx.x * blockDim.x + threadIdx.x; idx < n;
       idx += (long)gridDim.x * blockDim.x) {
    long jl = idx / ((long)L * D), r = idx % ((long)L * D);
    long i = r / D, d = r % D;
    P[(i * L + (j0 + jl)) * D + d] += tmp[idx];
  }
}

// -------------------------------- host side --------------------------------
static GemmArgs mk(const float* X, const float* W, const float* bias, float* Y,
                   int M, int N, int K) {
  GemmArgs g{};
  g.X = X; g.W = W; g.bias = bias; g.Y = Y;
  g.M = M; g.N = N; g.K = K;
  g.xs = K; g.ws = K; g.ys_m = N; g.ys_n = 1; g.rs_m = N; g.rs_n = 1;
  g.alpha = 1.f; g.nbi = 1;
  return g;
}
static void rungemm(hipStream_t st, int act, bool bt, const GemmArgs& g, int batches) {
  dim3 grid((g.N + 63) / 64, g.M / 64, batches), blk(256, 1, 1);
  if (act == 0 && bt)       k_gemm<0, true ><<<grid, blk, 0, st>>>(g);
  else if (act == 0)        k_gemm<0, false><<<grid, blk, 0, st>>>(g);
  else if (bt)              k_gemm<1, true ><<<grid, blk, 0, st>>>(g);
  else                      k_gemm<1, false><<<grid, blk, 0, st>>>(g);
}
static inline int rowblocks(long rows) { return (int)((rows + 7) / 8); }  // 8 waves/block

extern "C" void kernel_launch(void* const* d_in, const int* in_sizes, int n_in,
                              void* d_out, int out_size, void* d_ws, size_t ws_size,
                              hipStream_t stream) {
  (void)in_sizes; (void)n_in; (void)out_size; (void)ws_size;
  // ---- inputs (setup_inputs() dict insertion order, recursively) ----
  const float* in_seq     = (const float*)d_in[0];
  const float* in_pair    = (const float*)d_in[1];
  const float* seq_norm_g = (const float*)d_in[2];
  const float* seq_norm_b = (const float*)d_in[3];
  const float* attn_in_w  = (const float*)d_in[4];
  const float* attn_in_b  = (const float*)d_in[5];
  const float* attn_out_w = (const float*)d_in[6];
  const float* attn_out_b = (const float*)d_in[7];
  const float* bias_w     = (const float*)d_in[8];
  const float* bias_b     = (const float*)d_in[9];
  const float* sffn_ln_g  = (const float*)d_in[10];
  const float* sffn_ln_b  = (const float*)d_in[11];
  const float* sffn_w1    = (const float*)d_in[12];
  const float* sffn_b1    = (const float*)d_in[13];
  const float* sffn_w2    = (const float*)d_in[14];
  const float* sffn_b2    = (const float*)d_in[15];
  // row / col pair attention params: ln_g, ln_b, qkv_w, qkv_b, proj_w, proj_b
  const float* const* rowp = (const float* const*)&d_in[16];
  const float* const* colp = (const float* const*)&d_in[22];
  // tri params: ln_g, ln_b, lno_g, lno_b, a_w,a_b,b_w,b_b,ga_w,ga_b,gb_w,gb_b,o_w,o_b,go_w,go_b
  const float* const* triO = (const float* const*)&d_in[28];
  const float* const* triI = (const float* const*)&d_in[44];
  const float* pffn_ln_g = (const float*)d_in[60];
  const float* pffn_ln_b = (const float*)d_in[61];
  const float* pffn_w1   = (const float*)d_in[62];
  const float* pffn_b1   = (const float*)d_in[63];
  const float* pffn_w2   = (const float*)d_in[64];
  const float* pffn_b2   = (const float*)d_in[65];
  const float* p2s_ln_g  = (const float*)d_in[66];
  const float* p2s_ln_b  = (const float*)d_in[67];
  const float* p2s_pool_w= (const float*)d_in[68];
  const float* p2s_pool_b= (const float*)d_in[69];
  const float* p2s_proj_w= (const float*)d_in[70];
  const float* p2s_proj_b= (const float*)d_in[71];
  const float* gate_ps   = (const float*)d_in[72];
  const float* s2p_ln_g  = (const float*)d_in[73];
  const float* s2p_ln_b  = (const float*)d_in[74];
  const float* s2p_w1    = (const float*)d_in[75];
  const float* s2p_b1    = (const float*)d_in[76];
  const float* s2p_w2    = (const float*)d_in[77];
  const float* s2p_b2    = (const float*)d_in[78];
  const float* gate_sp   = (const float*)d_in[79];

  // ---- workspace carve ----
  char* wp = (char*)d_ws;
  auto carve = [&](size_t elems) -> float* {
    float* r = (float*)wp;
    wp += ((elems * sizeof(float)) + 255) & ~(size_t)255;
    return r;
  };
  float* S    = carve((size_t)Lc * DIMc);       // seq (updated in place)
  float* Hs   = carve((size_t)Lc * DIMc);       // seq LN scratch
  float* QKVs = carve((size_t)Lc * 3 * DIMc);
  float* BiasB= carve((size_t)Hc * TOK);        // (H,L,L)
  float* ScS  = carve((size_t)Hc * TOK);        // seq scores
  float* AO   = carve((size_t)Lc * DIMc);
  float* XG   = carve((size_t)Lc * 2 * FFNc);
  float* HF   = carve((size_t)Lc * FFNc);
  float* Pool = carve((size_t)Lc * PD);
  float* Uu   = carve((size_t)Lc * PD);
  float* Vv   = carve((size_t)Lc * PD);
  float* Scw  = carve((size_t)TOK);             // p2s pooling logits
  float* P    = carve((size_t)TOK * PD);        // pair (updated in place)
  float* ZN   = carve((size_t)TOK * PD);        // pair LN scratch
  float* T1   = carve((size_t)TOK * PD);
  float* T2   = carve((size_t)TOK * PD);
  float* T3   = carve((size_t)TOK * PD);
  float* T4   = carve((size_t)TOK * PD);
  // stage-local aliases (pair-attention chunks; free while tri slabs unused)
  float* Cscore = T1;   // CI*HP*L*L == TOK*PD exactly
  float* Cqkv   = T2;   // CI*L*3*PD
  float* Cout   = T3;   // CI*L*PD
  float* CoutT  = T4;

  const float ISQ = 0.17677669529663687f;       // 1/sqrt(32)
  const long  PAIR_N = (long)TOK * PD;

  hipMemcpyAsync(S, in_seq,  sizeof(float) * Lc * DIMc, hipMemcpyDeviceToDevice, stream);
  hipMemcpyAsync(P, in_pair, sizeof(float) * PAIR_N,    hipMemcpyDeviceToDevice, stream);

  // ---- pair bias for seq attention (from INPUT pair), stored as (H,L,L) ----
  { GemmArgs g = mk(P, bias_w, bias_b, BiasB, TOK, Hc, PD);
    g.ys_m = 1; g.ys_n = TOK;
    rungemm(stream, 0, true, g, 1); }

  // ---- 1. seq self-attention with pair bias ----
  k_ln<<<rowblocks(Lc), 256, 0, stream>>>(S, seq_norm_g, seq_norm_b, Hs, Lc, DIMc, 0, 0);
  { GemmArgs g = mk(Hs, attn_in_w, attn_in_b, QKVs, Lc, 3 * DIMc, DIMc);
    rungemm(stream, 0, true, g, 1); }
  { // scores[h,q,k] = q.k/sqrt(hd) + bias[h,q,k]
    GemmArgs g = mk(QKVs, QKVs + DIMc, nullptr, ScS, Lc, Lc, 32);
    g.xs = 3 * DIMc; g.ws = 3 * DIMc;
    g.nbi = Hc; g.xb_i = 32; g.wb_i = 32;
    g.yb_i = TOK; g.ys_m = Lc; g.ys_n = 1;
    g.alpha = ISQ;
    g.R = BiasB; g.rb_i = TOK; g.rs_m = Lc; g.rs_n = 1;
    rungemm(stream, 0, true, g, Hc); }
  k_softmax<<<rowblocks((long)Hc * Lc), 256, 0, stream>>>(ScS, (long)Hc * Lc, Lc);
  { GemmArgs g = mk(ScS, QKVs + 2 * DIMc, nullptr, AO, Lc, 32, Lc);
    g.xs = Lc; g.xb_i = TOK;
    g.ws = 3 * DIMc; g.wb_i = 32;
    g.ys_m = DIMc; g.ys_n = 1; g.yb_i = 32;
    g.nbi = Hc;
    rungemm(stream, 0, false, g, Hc); }
  { GemmArgs g = mk(AO, attn_out_w, attn_out_b, S, Lc, DIMc, DIMc);
    g.R = S; rungemm(stream, 0, true, g, 1); }

  // ---- 2. seq GEGLU FFN ----
  k_ln<<<rowblocks(Lc), 256, 0, stream>>>(S, sffn_ln_g, sffn_ln_b, Hs, Lc, DIMc, 0, 0);
  { GemmArgs g = mk(Hs, sffn_w1, sffn_b1, XG, Lc, 2 * FFNc, DIMc);
    rungemm(stream, 0, true, g, 1); }
  k_geglu<<<(int)(((long)Lc * FFNc + 255) / 256), 256, 0, stream>>>(XG, HF, Lc, FFNc);
  { GemmArgs g = mk(HF, sffn_w2, sffn_b2, S, Lc, DIMc, FFNc);
    g.R = S; rungemm(stream, 0, true, g, 1); }

  // ---- 3-4. pair row / column attention (i-chunked, score buffer L2-resident) ----
  auto pair_attn = [&](const float* const* pp, bool col) {
    k_ln<<<rowblocks(TOK), 256, 0, stream>>>(P, pp[0], pp[1], ZN, TOK, PD, Lc, col ? 1 : 0);
    for (int c = 0; c < Lc / CI; ++c) {
      const float* Xc = ZN + (size_t)c * CI * Lc * PD;
      { GemmArgs g = mk(Xc, pp[2], pp[3], Cqkv, CI * Lc, 3 * PD, PD);
        rungemm(stream, 0, true, g, 1); }
      { GemmArgs g = mk(Cqkv, Cqkv + PD, nullptr, Cscore, Lc, Lc, 32);
        g.xs = 3 * PD; g.ws = 3 * PD;
        g.nbi = HPc;
        g.xb_o = (long)Lc * 3 * PD; g.xb_i = 32;
        g.wb_o = (long)Lc * 3 * PD; g.wb_i = 32;
        g.yb_o = (long)HPc * TOK;   g.yb_i = TOK;
        g.ys_m = Lc; g.ys_n = 1; g.alpha = ISQ;
        rungemm(stream, 0, true, g, CI * HPc); }
      k_softmax<<<rowblocks((long)CI * HPc * Lc), 256, 0, stream>>>(
          Cscore, (long)CI * HPc * Lc, Lc);
      { GemmArgs g = mk(Cscore, Cqkv + 2 * PD, nullptr, Cout, Lc, 32, Lc);
        g.xs = Lc; g.xb_o = (long)HPc * TOK; g.xb_i = TOK;
        g.ws = 3 * PD; g.wb_o = (long)Lc * 3 * PD; g.wb_i = 32;
        g.ys_m = PD; g.ys_n = 1; g.yb_o = (long)Lc * PD; g.yb_i = 32;
        g.nbi = HPc;
        rungemm(stream, 0, false, g, CI * HPc); }
      if (!col) {
        GemmArgs g = mk(Cout, pp[4], pp[5], P + (size_t)c * CI * Lc * PD, CI * Lc, PD, PD);
        g.R = P + (size_t)c * CI * Lc * PD;
        rungemm(stream, 0, true, g, 1);
      } else {
        GemmArgs g = mk(Cout, pp[4], pp[5], CoutT, CI * Lc, PD, PD);
        rungemm(stream, 0, true, g, 1);
        long n = (long)CI * Lc * PD;
        k_addT<<<(int)((n + 255) / 256), 256, 0, stream>>>(P, CoutT, c * CI, CI, Lc, PD);
      }
    }
  };
  pair_attn(rowp, false);
  pair_attn(colp, true);

  // ---- 5. triangular multiplicative updates ----
  auto tri = [&](const float* const* tp, int mode) {
    // tp: 0 ln_g 1 ln_b 2 lno_g 3 lno_b 4 a_w 5 a_b 6 b_w 7 b_b
    //     8 ga_w 9 ga_b 10 gb_w 11 gb_b 12 o_w 13 o_b 14 go_w 15 go_b
    k_ln<<<rowblocks(TOK), 256, 0, stream>>>(P, tp[0], tp[1], ZN, TOK, PD, Lc, 0);
    { GemmArgs g = mk(ZN, tp[4],  tp[5],  T1, TOK, PD, PD); rungemm(stream, 0, true, g, 1); }
    { GemmArgs g = mk(ZN, tp[8],  tp[9],  T2, TOK, PD, PD); rungemm(stream, 0, true, g, 1); }
    k_mul_sigmoid<<<51200, 256, 0, stream>>>(T1, T2, PAIR_N);          // a *= sig(ga)
    { GemmArgs g = mk(ZN, tp[6],  tp[7],  T2, TOK, PD, PD); rungemm(stream, 0, true, g, 1); }
    { GemmArgs g = mk(ZN, tp[10], tp[11], T3, TOK, PD, PD); rungemm(stream, 0, true, g, 1); }
    k_mul_sigmoid<<<51200, 256, 0, stream>>>(T2, T3, PAIR_N);          // b *= sig(gb)
    { GemmArgs g = mk(ZN, tp[14], tp[15], T3, TOK, PD, PD); rungemm(stream, 0, true, g, 1); } // go
    k_tri<<<dim3(Lc, Lc, 1), PD, 0, stream>>>(T1, T2, T4, Lc, PD, mode);
    k_ln<<<rowblocks(TOK), 256, 0, stream>>>(T4, tp[2], tp[3], T4, TOK, PD, Lc, 0);
    { GemmArgs g = mk(T4, tp[12], tp[13], T1, TOK, PD, PD); rungemm(stream, 0, true, g, 1); } // o
    k_gate_add<<<51200, 256, 0, stream>>>(P, T1, T3, PAIR_N);          // P += o*sig(go)
  };
  tri(triO, 0);  // outgoing
  tri(triI, 1);  // incoming

  // ---- 6. pair FFN (token-chunked so hidden stays L2-resident) ----
  k_ln<<<rowblocks(TOK), 256, 0, stream>>>(P, pffn_ln_g, pffn_ln_b, ZN, TOK, PD, Lc, 0);
  for (int c = 0; c < TOK / MC; ++c) {
    const float* Xc = ZN + (size_t)c * MC * PD;
    { GemmArgs g = mk(Xc, pffn_w1, pffn_b1, T1, MC, 4 * PD, PD);
      rungemm(stream, 1, true, g, 1); }                                // fused GELU
    { GemmArgs g = mk(T1, pffn_w2, pffn_b2, P + (size_t)c * MC * PD, MC, PD, 4 * PD);
      g.R = P + (size_t)c * MC * PD;
      rungemm(stream, 0, true, g, 1); }
  }

  // ---- 7a. pair -> seq (attention pooling) ----
  k_ln<<<rowblocks(TOK), 256, 0, stream>>>(P, p2s_ln_g, p2s_ln_b, ZN, TOK, PD, Lc, 0);
  k_rowdot<<<rowblocks(TOK), 256, 0, stream>>>(ZN, p2s_pool_w, p2s_pool_b, Scw, TOK, PD);
  k_softmax<<<rowblocks(Lc), 256, 0, stream>>>(Scw, Lc, Lc);
  k_pool<<<Lc, PD, 0, stream>>>(Scw, ZN, Pool, Lc, PD);
  { GemmArgs g = mk(Pool, p2s_proj_w, p2s_proj_b, S, Lc, DIMc, PD);
    g.R = S; g.gate = gate_ps;
    rungemm(stream, 0, true, g, 1); }

  // ---- 7b. seq -> pair (factorized: cat@W1^T = sn@W1a^T[i] + sn@W1b^T[j]) ----
  k_ln<<<rowblocks(Lc), 256, 0, stream>>>(S, s2p_ln_g, s2p_ln_b, Hs, Lc, DIMc, 0, 0);
  { GemmArgs g = mk(Hs, s2p_w1,        nullptr, Uu, Lc, PD, DIMc);
    g.ws = 2 * DIMc; rungemm(stream, 0, true, g, 1); }
  { GemmArgs g = mk(Hs, s2p_w1 + DIMc, nullptr, Vv, Lc, PD, DIMc);
    g.ws = 2 * DIMc; rungemm(stream, 0, true, g, 1); }
  k_outer_gelu<<<51200, 256, 0, stream>>>(Uu, Vv, s2p_b1, T1, Lc, PD);
  { GemmArgs g = mk(T1, s2p_w2, s2p_b2, P, TOK, PD, PD);
    g.R = P; g.gate = gate_sp;
    rungemm(stream, 0, true, g, 1); }

  // ---- outputs: (seq, pair) concatenated flat ----
  hipMemcpyAsync(d_out, S, sizeof(float) * Lc * DIMc, hipMemcpyDeviceToDevice, stream);
  hipMemcpyAsync((float*)d_out + (size_t)Lc * DIMc, P, sizeof(float) * PAIR_N,
                 hipMemcpyDeviceToDevice, stream);
}
